// TransformerBlock_40819369181947
// MI455X (gfx1250) — matrix operand order repaired and checked by reading, not compiled, not run on hardware
//
#include <hip/hip_runtime.h>
#include <math.h>

typedef _Float16 f16;
typedef __attribute__((ext_vector_type(16))) _Float16 v16h;
typedef __attribute__((ext_vector_type(8)))  _Float16 v8h;
typedef __attribute__((ext_vector_type(8)))  float    v8f;

// ---------------------------------------------------------------------------
// Fragment load helpers (CDNA5 WMMA 16x16x32 f16 layouts, wave32)
// A 16x32: lane l -> row = l%16 ; elems 0..7 = K g*8..g*8+7, elems 8..15 = K 16+g*8.. (g=l/16)
// B 32x16: lane l -> col = l%16 ; elems 0..7 = K g*8..g*8+7, elems 8..15 = K 16+g*8.. (g=l/16)
// C 16x16: lane l, vgpr r -> row = r + 8*(l/16), col = l%16
// ---------------------------------------------------------------------------
__device__ __forceinline__ v16h ld_frag_a(const f16* p) { // p = &row[g*8]
  v8h x = *(const v8h*)p;
  v8h y = *(const v8h*)(p + 16);
  v16h r;
#pragma unroll
  for (int i = 0; i < 8; ++i) { r[i] = x[i]; r[i + 8] = y[i]; }
  return r;
}
__device__ __forceinline__ v16h ld_frag_b(const f16* p, int pitch) { // p = &tile[g*8][n0+l%16], K strided by pitch
  v16h r;
#pragma unroll
  for (int i = 0; i < 8; ++i) { r[i] = p[i * pitch]; r[i + 8] = p[(16 + i) * pitch]; }
  return r;
}
__device__ __forceinline__ v8f wmma32(v16h a, v16h b, v8f c) {
  return __builtin_amdgcn_wmma_f32_16x16x32_f16(false, a, false, b, (short)0, c, false, false);
}

// Async global -> LDS copy of 16 bytes per lane (CDNA5, tracked by ASYNCcnt).
// LDS byte offset == low 32 bits of the generic shared-space pointer.
__device__ __forceinline__ void async_ld_lds16(const void* gaddr, void* lds) {
  unsigned int la = (unsigned int)(unsigned long long)lds;
  asm volatile("global_load_async_to_lds_b128 %0, %1, off"
               :: "v"(la), "v"(gaddr) : "memory");
}
__device__ __forceinline__ void wait_async0() {
  asm volatile("s_wait_asynccnt 0x0" ::: "memory");
}
__device__ __forceinline__ void wait_async3() {
  asm volatile("s_wait_asynccnt 0x3" ::: "memory");
}

// ---------------------------------------------------------------------------
// Elementwise / prep kernels
// ---------------------------------------------------------------------------
__global__ void cast_f32_to_f16(const float* __restrict__ s, f16* __restrict__ d, size_t n) {
  size_t stride = (size_t)gridDim.x * blockDim.x;
  for (size_t i = (size_t)blockIdx.x * blockDim.x + threadIdx.x; i < n; i += stride)
    d[i] = (f16)s[i];
}

// copy src [rows,scols] fp32 into f16 dst at column offset inside pitch dpitch
__global__ void cast_cols_f16(const float* __restrict__ s, f16* __restrict__ d,
                              int rows, int scols, int dpitch, int coff) {
  size_t n = (size_t)rows * scols;
  size_t stride = (size_t)gridDim.x * blockDim.x;
  for (size_t i = (size_t)blockIdx.x * blockDim.x + threadIdx.x; i < n; i += stride) {
    int r = (int)(i / scols), c = (int)(i % scols);
    d[(size_t)r * dpitch + coff + c] = (f16)s[i];
  }
}

__global__ void copy_f32(const float* __restrict__ s, float* __restrict__ d, int n) {
  int i = blockIdx.x * blockDim.x + threadIdx.x;
  if (i < n) d[i] = s[i];
}

__global__ void __launch_bounds__(256)
rmsnorm_cast_kernel(const float* __restrict__ x, const float* __restrict__ w,
                    f16* __restrict__ out, int cols) {
  int row = blockIdx.x;
  const float* xr = x + (size_t)row * cols;
  f16* orow = out + (size_t)row * cols;
  float ss = 0.f;
  for (int c = threadIdx.x; c < cols; c += 256) { float v = xr[c]; ss += v * v; }
#pragma unroll
  for (int off = 16; off > 0; off >>= 1) ss += __shfl_xor(ss, off, 32);
  __shared__ float red[8];
  int wv = threadIdx.x >> 5, ln = threadIdx.x & 31;
  if (ln == 0) red[wv] = ss;
  __syncthreads();
  if (threadIdx.x == 0) {
    float t = 0.f;
#pragma unroll
    for (int i = 0; i < 8; ++i) t += red[i];
    red[0] = rsqrtf(t / (float)cols + 1.1920929e-07f);
  }
  __syncthreads();
  float sc = red[0];
  for (int c = threadIdx.x; c < cols; c += 256) orow[c] = (f16)(xr[c] * sc * w[c]);
}

// ---------------------------------------------------------------------------
// WMMA GEMM: C[M,N] = act(A[M,K]f16 @ B[K,N]f16 + bias) (+resid) -> f32/f16
// 256 thr = 8 waves (2M x 4N), block tile 64x128, each wave 32x32, K step 32.
// Double-buffered LDS staging via global_load_async_to_lds_b128 (ASYNCcnt).
// ---------------------------------------------------------------------------
#define GM 64
#define GN 128
#define APITCH 40
#define BPITCH 136

__device__ __forceinline__ float gelu_exact(float v) {
  return 0.5f * v * (1.0f + erff(v * 0.70710678118654752f));
}

template <int ACT>
__global__ void __launch_bounds__(256)
gemm_wmma(const f16* __restrict__ A, const f16* __restrict__ Bw,
          const float* __restrict__ bias, const float* __restrict__ resid,
          float* __restrict__ outF, f16* __restrict__ outH,
          int M, int N, int Kd) {
  __shared__ f16 sA[2][GM][APITCH];
  __shared__ f16 sB[2][32][BPITCH];
  int tid = threadIdx.x, lane = tid & 31, w = tid >> 5;
  int wm = w & 1, wn = w >> 1;
  int lr = lane & 15, g = lane >> 4;
  int bm = blockIdx.y * GM;
  int bn = blockIdx.x * GN;

  int arow = tid >> 2, acol = (tid & 3) * 8;   // A tile: 64 rows x 4 thr x 8 f16
  int brow = tid >> 3, bcol = (tid & 7) * 16;  // B tile: 32 rows x 8 thr x 16 f16

  const f16* gA = A + (size_t)(bm + arow) * Kd + acol;
  const f16* gB = Bw + (size_t)brow * N + bn + bcol;

  v8f c[2][2];
#pragma unroll
  for (int mi = 0; mi < 2; ++mi)
#pragma unroll
    for (int ni = 0; ni < 2; ++ni) { v8f z = {}; c[mi][ni] = z; }

  // prologue: stage tile 0 into buffer 0
  async_ld_lds16(gA, &sA[0][arow][acol]);
  async_ld_lds16(gB, &sB[0][brow][bcol]);
  async_ld_lds16(gB + 8, &sB[0][brow][bcol + 8]);

  int nk = Kd >> 5;
  for (int kt = 0; kt < nk; ++kt) {
    int buf = kt & 1;
    if (kt + 1 < nk) {
      // stage next tile into the other buffer while this one is consumed
      const f16* nA = gA + (kt + 1) * 32;
      const f16* nB = gB + (size_t)(kt + 1) * 32 * N;
      async_ld_lds16(nA, &sA[buf ^ 1][arow][acol]);
      async_ld_lds16(nB, &sB[buf ^ 1][brow][bcol]);
      async_ld_lds16(nB + 8, &sB[buf ^ 1][brow][bcol + 8]);
      wait_async3();   // in-order completion: current tile's 3 loads have landed
    } else {
      wait_async0();
    }
    __syncthreads();

    v16h a[2];
#pragma unroll
    for (int mi = 0; mi < 2; ++mi)
      a[mi] = ld_frag_a(&sA[buf][wm * 32 + mi * 16 + lr][g * 8]);
#pragma unroll
    for (int ni = 0; ni < 2; ++ni) {
      v16h b = ld_frag_b(&sB[buf][g * 8][wn * 32 + ni * 16 + lr], BPITCH);
#pragma unroll
      for (int mi = 0; mi < 2; ++mi)
        c[mi][ni] = wmma32(a[mi], b, c[mi][ni]);
    }
    __syncthreads();  // all waves done reading buf before it is overwritten
  }

#pragma unroll
  for (int mi = 0; mi < 2; ++mi)
#pragma unroll
    for (int ni = 0; ni < 2; ++ni) {
      int col = bn + wn * 32 + ni * 16 + lr;
      float bv = bias[col];
#pragma unroll
      for (int r = 0; r < 8; ++r) {
        int row = bm + wm * 32 + mi * 16 + r + 8 * g;
        float v = c[mi][ni][r] + bv;
        if (ACT == 1) v = gelu_exact(v);
        size_t idx = (size_t)row * N + col;
        if (resid) v += resid[idx];
        if (outF) outF[idx] = v;
        if (outH) outH[idx] = (f16)v;
      }
    }
}

// ---------------------------------------------------------------------------
// Flash attention: grid (S/128, H, B), 128 thr = 4 waves, wave owns 32 q rows
// qkv: [B*S, 3E] f16 (q | k | v). Online softmax, WMMA QK^T and PV.
// V tile staged via async-to-LDS; K tile staged transposed via VGPR scatter.
// ---------------------------------------------------------------------------
#define FAD 64
#define KTP 40
#define VP  72
#define PP  40

__global__ void __launch_bounds__(128)
flash_attn_kernel(const f16* __restrict__ qkv, const int* __restrict__ mask,
                  f16* __restrict__ attnout, int S, int E) {
  const int QK = 3 * E;
  int h = blockIdx.y, b = blockIdx.z;
  int tid = threadIdx.x, lane = tid & 31, w = tid >> 5;
  int lr = lane & 15, g = lane >> 4;

  size_t base = (size_t)b * S * QK;
  const f16* Q = qkv + base + h * FAD;
  const f16* K = qkv + base + E + h * FAD;
  const f16* V = qkv + base + 2 * E + h * FAD;

  __shared__ f16 sKT[FAD][KTP];      // [d][key]
  __shared__ f16 sV[32][VP];         // [key][d]
  __shared__ f16 sP[4][32][PP];      // per-wave prob tile

  int q0 = blockIdx.x * 128 + w * 32;

  // resident Q A-fragments: [mi][dchunk]
  v16h qa[2][2];
#pragma unroll
  for (int mi = 0; mi < 2; ++mi)
#pragma unroll
    for (int dc = 0; dc < 2; ++dc)
      qa[mi][dc] = ld_frag_a(Q + (size_t)(q0 + mi * 16 + lr) * QK + dc * 32 + g * 8);

  float m_i[2][8], l_i[2][8];
  v8f o[2][4];
#pragma unroll
  for (int mi = 0; mi < 2; ++mi) {
#pragma unroll
    for (int r = 0; r < 8; ++r) { m_i[mi][r] = -3.0e38f; l_i[mi][r] = 0.f; }
#pragma unroll
    for (int ni = 0; ni < 4; ++ni) { v8f z = {}; o[mi][ni] = z; }
  }

  const float scale = 0.125f; // 1/sqrt(64)

  for (int kb = 0; kb < S; kb += 32) {
    __syncthreads();
    { // stage: 32 keys x 64 d. V via async DMA to LDS; K via VGPR transpose.
      int key = tid >> 2;
      int dp  = (tid & 3) * 16;
      const f16* vp = V + (size_t)(kb + key) * QK + dp;
      async_ld_lds16(vp, &sV[key][dp]);
      async_ld_lds16(vp + 8, &sV[key][dp + 8]);
      const f16* kp = K + (size_t)(kb + key) * QK + dp;
      v8h k0v = *(const v8h*)kp;
      v8h k1v = *(const v8h*)(kp + 8);
#pragma unroll
      for (int i = 0; i < 8; ++i) { sKT[dp + i][key] = k0v[i]; sKT[dp + 8 + i][key] = k1v[i]; }
    }
    wait_async0();
    __syncthreads();

    // scores S = Q K^T * scale  (2x2 fragments of 16x16 over 32 q x 32 keys)
    v8f s[2][2];
#pragma unroll
    for (int ni = 0; ni < 2; ++ni) {
      v16h bk0 = ld_frag_b(&sKT[0 * 32 + g * 8][ni * 16 + lr], KTP);
      v16h bk1 = ld_frag_b(&sKT[1 * 32 + g * 8][ni * 16 + lr], KTP);
#pragma unroll
      for (int mi = 0; mi < 2; ++mi) {
        v8f acc = {};
        acc = wmma32(qa[mi][0], bk0, acc);
        acc = wmma32(qa[mi][1], bk1, acc);
#pragma unroll
        for (int r = 0; r < 8; ++r) acc[r] *= scale;
        s[mi][ni] = acc;
      }
    }
    // mask (mask[b, key] == 0 -> -inf)
#pragma unroll
    for (int ni = 0; ni < 2; ++ni) {
      int mk = mask[(size_t)b * S + kb + ni * 16 + lr];
      if (mk == 0) {
#pragma unroll
        for (int mi = 0; mi < 2; ++mi)
#pragma unroll
          for (int r = 0; r < 8; ++r) s[mi][ni][r] = -1.0e30f;
      }
    }

    // online softmax: lane's rows are r + 8*g (+16*mi); reduce across 16-lane half
#pragma unroll
    for (int mi = 0; mi < 2; ++mi) {
#pragma unroll
      for (int r = 0; r < 8; ++r) {
        float mx = fmaxf(s[mi][0][r], s[mi][1][r]);
#pragma unroll
        for (int off = 1; off < 16; off <<= 1) mx = fmaxf(mx, __shfl_xor(mx, off, 32));
        float mnew = fmaxf(m_i[mi][r], mx);
        float corr = __expf(m_i[mi][r] - mnew);
        float p0 = __expf(s[mi][0][r] - mnew);
        float p1 = __expf(s[mi][1][r] - mnew);
        s[mi][0][r] = p0; s[mi][1][r] = p1;
        float rs = p0 + p1;
#pragma unroll
        for (int off = 1; off < 16; off <<= 1) rs += __shfl_xor(rs, off, 32);
        l_i[mi][r] = l_i[mi][r] * corr + rs;
        m_i[mi][r] = mnew;
#pragma unroll
        for (int ni = 0; ni < 4; ++ni) o[mi][ni][r] *= corr;
      }
    }

    // C-layout P -> LDS -> A-layout (per-wave tile; barrier keeps waves lockstep)
#pragma unroll
    for (int mi = 0; mi < 2; ++mi)
#pragma unroll
      for (int ni = 0; ni < 2; ++ni)
#pragma unroll
        for (int r = 0; r < 8; ++r)
          sP[w][mi * 16 + r + 8 * g][ni * 16 + lr] = (f16)s[mi][ni][r];
    __syncthreads();

    // O += P V
    v16h pa[2];
#pragma unroll
    for (int mi = 0; mi < 2; ++mi)
      pa[mi] = ld_frag_a(&sP[w][mi * 16 + lr][g * 8]);
#pragma unroll
    for (int ni = 0; ni < 4; ++ni) {
      v16h vb = ld_frag_b(&sV[g * 8][ni * 16 + lr], VP);
#pragma unroll
      for (int mi = 0; mi < 2; ++mi)
        o[mi][ni] = wmma32(pa[mi], vb, o[mi][ni]);
    }
  }

  // finalize: divide by l, write f16 [B*S, E] with head offset
#pragma unroll
  for (int mi = 0; mi < 2; ++mi)
#pragma unroll
    for (int r = 0; r < 8; ++r) {
      float inv = 1.0f / l_i[mi][r];
      int row = q0 + mi * 16 + r + 8 * g;
#pragma unroll
      for (int ni = 0; ni < 4; ++ni) {
        int col = h * FAD + ni * 16 + lr;
        attnout[(size_t)((size_t)b * S + row) * E + col] = (f16)(o[mi][ni][r] * inv);
      }
    }
}

// ---------------------------------------------------------------------------
// Host orchestration
// ---------------------------------------------------------------------------
extern "C" void kernel_launch(void* const* d_in, const int* in_sizes, int n_in,
                              void* d_out, int out_size, void* d_ws, size_t ws_size,
                              hipStream_t stream) {
  const int Bb = 4, S = 2048, E = 1024, Hh = 16, FF = 3072;
  const int M = Bb * S;

  const float* x    = (const float*)d_in[0];
  const int*   mask = (const int*)d_in[1];
  const float* Wq   = (const float*)d_in[2];
  const float* bq   = (const float*)d_in[3];
  const float* Wkv  = (const float*)d_in[4];
  const float* bkv  = (const float*)d_in[5];
  const float* Wo   = (const float*)d_in[6];
  const float* bo   = (const float*)d_in[7];
  const float* n1   = (const float*)d_in[8];
  const float* n3   = (const float*)d_in[9];
  const float* W1   = (const float*)d_in[10];
  const float* b1   = (const float*)d_in[11];
  const float* W2   = (const float*)d_in[12];
  const float* b2   = (const float*)d_in[13];

  char* ws = (char*)d_ws;
  size_t off = 0;
  auto take = [&](size_t bytes) -> char* {
    char* p = ws + off;
    off = (off + bytes + 255) & ~(size_t)255;
    return p;
  };

  f16*   wqkv    = (f16*)take((size_t)E * 3 * E * 2);
  float* bqkv    = (float*)take((size_t)3 * E * 4);
  f16*   wo16    = (f16*)take((size_t)E * E * 2);
  f16*   w116    = (f16*)take((size_t)E * FF * 2);
  f16*   w216    = (f16*)take((size_t)FF * E * 2);
  f16*   xn      = (f16*)take((size_t)M * E * 2);
  f16*   qkv     = (f16*)take((size_t)M * 3 * E * 2);
  f16*   attnout = (f16*)take((size_t)M * E * 2);
  float* x2      = (float*)take((size_t)M * E * 4);
  f16*   hmid    = qkv; // reuse (qkv dead after attention)
  f16*   xn2     = xn;  // reuse

  // weight prep (f16, fused QKV)
  cast_cols_f16<<<512, 256, 0, stream>>>(Wq, wqkv, E, E, 3 * E, 0);
  cast_cols_f16<<<1024, 256, 0, stream>>>(Wkv, wqkv, E, 2 * E, 3 * E, E);
  copy_f32<<<(E + 255) / 256, 256, 0, stream>>>(bq, bqkv, E);
  copy_f32<<<(2 * E + 255) / 256, 256, 0, stream>>>(bkv, bqkv + E, 2 * E);
  cast_f32_to_f16<<<512, 256, 0, stream>>>(Wo, wo16, (size_t)E * E);
  cast_f32_to_f16<<<1024, 256, 0, stream>>>(W1, w116, (size_t)E * FF);
  cast_f32_to_f16<<<1024, 256, 0, stream>>>(W2, w216, (size_t)FF * E);

  // 1) xn = rmsnorm(x, norm1) -> f16
  rmsnorm_cast_kernel<<<M, 256, 0, stream>>>(x, n1, xn, E);

  // 2) qkv = xn @ [Wq|Wkv] + [bq|bkv]  (f16 out)
  {
    dim3 grid(3 * E / GN, M / GM);
    gemm_wmma<0><<<grid, 256, 0, stream>>>(xn, wqkv, bqkv, nullptr, nullptr, qkv,
                                           M, 3 * E, E);
  }

  // 3) attention (flash, per (b,h,128 q rows))
  {
    dim3 grid(S / 128, Hh, Bb);
    flash_attn_kernel<<<grid, 128, 0, stream>>>(qkv, mask, attnout, S, E);
  }

  // 4) x2 = x + attnout @ Wo + bo  (f32 out)
  {
    dim3 grid(E / GN, M / GM);
    gemm_wmma<0><<<grid, 256, 0, stream>>>(attnout, wo16, bo, x, x2, nullptr,
                                           M, E, E);
  }

  // 5) xn2 = rmsnorm(x2, norm3) -> f16
  rmsnorm_cast_kernel<<<M, 256, 0, stream>>>(x2, n3, xn2, E);

  // 6) h = gelu(xn2 @ W1 + b1)  (f16 out)
  {
    dim3 grid(FF / GN, M / GM);
    gemm_wmma<1><<<grid, 256, 0, stream>>>(xn2, w116, b1, nullptr, nullptr, hmid,
                                           M, FF, E);
  }

  // 7) out = x2 + h @ W2 + b2  (f32 out)
  {
    dim3 grid(E / GN, M / GM);
    gemm_wmma<0><<<grid, 256, 0, stream>>>(hmid, w216, b2, x2, (float*)d_out, nullptr,
                                           M, E, FF);
  }
}
